// GeneralizedMamba2Naive_49598282334523
// MI455X (gfx1250) — matrix operand order repaired and checked by reading, not compiled, not run on hardware
//
#include <hip/hip_runtime.h>
#include <hip/hip_bf16.h>

// ---------------- problem constants ----------------
#define B_SZ     16
#define L_SEQ    512
#define DMODEL   1024
#define DSSM     2048
#define DSTATE   128
#define HEADDIM  64
#define NHEADS   32
#define DCONV    4
#define CONVDIM  2304              // DSSM + 2*DSTATE
#define DINPROJ  4384              // 2*DSSM + 2*DSTATE + NHEADS
#define NREC     2336              // CONVDIM + NHEADS (recurrent GEMM N)
#define EPSN     1e-5f

typedef __bf16 bf16_t;
typedef __attribute__((ext_vector_type(16))) __bf16 v16bf;
typedef __attribute__((ext_vector_type(8)))  float  v8f;

__device__ __forceinline__ float sigmoidf_(float x) { return 1.0f / (1.0f + __expf(-x)); }
__device__ __forceinline__ float siluf_(float x)    { return x * sigmoidf_(x); }
__device__ __forceinline__ float softplusf_(float x){ return (x > 20.0f) ? x : log1pf(__expf(x)); }

__device__ __forceinline__ unsigned short f32_to_bf16_bits(float f) {
    unsigned int u = __float_as_uint(f);
    u += 0x7FFFu + ((u >> 16) & 1u);   // round to nearest even
    return (unsigned short)(u >> 16);
}

// A-matrix per-lane fragment: one contiguous 16-element K-chunk of a row
__device__ __forceinline__ v16bf load_afrag(const bf16_t* q) {
    v16bf a;
    #pragma unroll
    for (int i = 0; i < 8; ++i) { a[i] = q[i]; a[8 + i] = q[8 + i]; }
    return a;
}
// B-matrix per-lane fragment: one contiguous 16-element K-chunk of a column
__device__ __forceinline__ v16bf load_bfrag(const bf16_t* q) {
    v16bf b;
    #pragma unroll
    for (int i = 0; i < 16; ++i) b[i] = q[i];
    return b;
}
#define WMMA_BF16(A, Bm, Cacc) \
    __builtin_amdgcn_wmma_f32_16x16x32_bf16(false, (A), false, (Bm), (short)0, (Cacc), false, false)

// ---------------- f32 -> bf16 conversion ----------------
__global__ void cvt_f32_bf16(const float* __restrict__ src,
                             unsigned short* __restrict__ dst, int n) {
    int i = blockIdx.x * blockDim.x + threadIdx.x;
    if (i < n) dst[i] = f32_to_bf16_bits(src[i]);
}

// ---------------- generic WMMA GEMM: C[M,N] = A[M,K] * B[N,K]^T ----------------
// Each wave: 32x32 output (2 M-tiles x 2 N-tiles), double-buffered K loop.
// Requires: M % 32 == 0, Ntiles even, K % 32 == 0.
__global__ __launch_bounds__(128) void wmma_gemm_bf16(
    const unsigned short* __restrict__ Abits, int lda,
    const unsigned short* __restrict__ Bbits, int ldb,
    float* __restrict__ C, int ldc, int K, int Ntiles)
{
    const int lane = threadIdx.x & 31;
    const int wave = threadIdx.x >> 5;
    const int mt0  = blockIdx.x * 2;
    const int nt0  = (blockIdx.y * 4 + wave) * 2;
    if (nt0 >= Ntiles) return;                       // wave-uniform

    const bf16_t* A = reinterpret_cast<const bf16_t*>(Abits);
    const bf16_t* B = reinterpret_cast<const bf16_t*>(Bbits);

    const int kbA = (lane < 16) ? 0 : 16;
    const int kbB = (lane < 16) ? 0 : 16;
    const bf16_t* pa0 = A + (long)(mt0 * 16 + (lane & 15)) * lda + kbA;
    const bf16_t* pa1 = pa0 + (long)16 * lda;
    const bf16_t* pb0 = B + (long)(nt0 * 16 + (lane & 15)) * ldb + kbB;
    const bf16_t* pb1 = pb0 + (long)16 * ldb;

    v8f c00 = {}, c01 = {}, c10 = {}, c11 = {};
    // prologue loads
    v16bf a0 = load_afrag(pa0), a1 = load_afrag(pa1);
    v16bf b0 = load_bfrag(pb0), b1 = load_bfrag(pb1);
    for (int k0 = 32; k0 < K; k0 += 32) {
        // issue next slice's loads before consuming current slice
        v16bf a0n = load_afrag(pa0 + k0), a1n = load_afrag(pa1 + k0);
        v16bf b0n = load_bfrag(pb0 + k0), b1n = load_bfrag(pb1 + k0);
        c00 = WMMA_BF16(a0, b0, c00);
        c01 = WMMA_BF16(a0, b1, c01);
        c10 = WMMA_BF16(a1, b0, c10);
        c11 = WMMA_BF16(a1, b1, c11);
        a0 = a0n; a1 = a1n; b0 = b0n; b1 = b1n;
    }
    c00 = WMMA_BF16(a0, b0, c00);
    c01 = WMMA_BF16(a0, b1, c01);
    c10 = WMMA_BF16(a1, b0, c10);
    c11 = WMMA_BF16(a1, b1, c11);

    const int n0 = nt0 * 16 + (lane & 15);
    const int mb = (lane < 16) ? 0 : 8;
    #pragma unroll
    for (int r = 0; r < 8; ++r) {
        long row0 = (long)(mt0 * 16 + mb + r) * ldc;
        long row1 = row0 + (long)16 * ldc;
        C[row0 + n0]      = c00[r];
        C[row0 + n0 + 16] = c01[r];
        C[row1 + n0]      = c10[r];
        C[row1 + n0 + 16] = c11[r];
    }
}

// ---------------- depthwise causal conv over L ----------------
__global__ void conv1d_kernel(const float* __restrict__ zx,
                              const float* __restrict__ conv_w,
                              const float* __restrict__ conv_b,
                              float* __restrict__ xbc_conv, int n)
{
    int idx = blockIdx.x * blockDim.x + threadIdx.x;
    if (idx >= n) return;
    int c  = idx % CONVDIM;
    int bl = idx / CONVDIM;
    int l  = bl % L_SEQ;
    int b  = bl / L_SEQ;
    float acc = conv_b[c];
    #pragma unroll
    for (int k = 0; k < DCONV; ++k) {
        int li = l + k - (DCONV - 1);
        if (li >= 0)
            acc += zx[((long)b * L_SEQ + li) * DINPROJ + DSSM + c] * conv_w[c * DCONV + k];
    }
    xbc_conv[((long)b * L_SEQ + l) * CONVDIM + c] = acc;
}

// ---------------- per-step recurrent WMMA GEMM + SiLU epilogue ----------------
// xbct[b,n] = silu(pre[b,n] + sum_k yn[b,t-1,k] * Wr[n,k]) , M = 16 (batch)
__global__ __launch_bounds__(128) void step_rec_gemm(
    const unsigned short* __restrict__ Ybits,   // yn_all bf16, offset to (t-1)*DSSM
    const unsigned short* __restrict__ Wrbits,  // 2336 x 2048 bf16
    const float* __restrict__ xbc_conv,
    const float* __restrict__ zx,
    float* __restrict__ xbct, int t, int first)
{
    const int lane = threadIdx.x & 31;
    const int wave = threadIdx.x >> 5;
    const int nt0  = (blockIdx.x * 4 + wave) * 2;
    const int Ntiles = NREC / 16;                 // 146 (even)
    if (nt0 >= Ntiles) return;                    // wave-uniform

    const bf16_t* A = reinterpret_cast<const bf16_t*>(Ybits);
    const bf16_t* W = reinterpret_cast<const bf16_t*>(Wrbits);
    const int lda = L_SEQ * DSSM;                 // batch stride inside yn_all

    const int kbA = (lane < 16) ? 0 : 16;
    const int kbB = (lane < 16) ? 0 : 16;
    const bf16_t* pa  = A + (long)(lane & 15) * lda + kbA;
    const bf16_t* pb0 = W + (long)(nt0 * 16 + (lane & 15)) * DSSM + kbB;
    const bf16_t* pb1 = pb0 + (long)16 * DSSM;

    v8f c0 = {}, c1 = {};
    if (!first) {
        v16bf a  = load_afrag(pa);
        v16bf b0 = load_bfrag(pb0), b1 = load_bfrag(pb1);
        for (int k0 = 32; k0 < DSSM; k0 += 32) {
            v16bf an  = load_afrag(pa + k0);
            v16bf b0n = load_bfrag(pb0 + k0), b1n = load_bfrag(pb1 + k0);
            c0 = WMMA_BF16(a, b0, c0);
            c1 = WMMA_BF16(a, b1, c1);
            a = an; b0 = b0n; b1 = b1n;
        }
        c0 = WMMA_BF16(a, b0, c0);
        c1 = WMMA_BF16(a, b1, c1);
    }
    const int n0 = nt0 * 16 + (lane & 15);
    const int mb = (lane < 16) ? 0 : 8;           // batch row base
    #pragma unroll
    for (int r = 0; r < 8; ++r) {
        int b = mb + r;                            // batch index (M = 16)
        #pragma unroll
        for (int j = 0; j < 2; ++j) {
            int n = n0 + j * 16;
            float pre = (n < CONVDIM)
                ? xbc_conv[((long)b * L_SEQ + t) * CONVDIM + n]
                : zx[((long)b * L_SEQ + t) * DINPROJ + DSSM + n]; // dt_raw region
            float v = (j == 0 ? c0[r] : c1[r]) + pre;
            xbct[b * NREC + n] = siluf_(v);
        }
    }
}

// ---------------- per-step SSM state update (L2-resident state) ----------------
__global__ __launch_bounds__(64) void step_state(
    const float* __restrict__ xbct, const float* __restrict__ zx,
    const float* __restrict__ dt_bias, const float* __restrict__ A_log,
    const float* __restrict__ D_param,
    float* __restrict__ state, float* __restrict__ g_buf, int t)
{
    const int h = blockIdx.x, b = blockIdx.y, p = threadIdx.x;
    __shared__ float sB[DSTATE], sC[DSTATE];
    const float* xb = xbct + b * NREC;
    for (int i = p; i < DSTATE; i += 64) {
        sB[i] = xb[DSSM + i];
        sC[i] = xb[DSSM + DSTATE + i];
    }
    __syncthreads();

    float dtr = xb[CONVDIM + h];                        // already silu'd
    float dt  = softplusf_(dtr + dt_bias[h]);
    float a   = -__expf(A_log[h]);
    float dA  = __expf(dt * a);
    float x   = xb[h * HEADDIM + p];
    float dtx = dt * x;

    float* st = state + (((long)(b * NHEADS + h) * HEADDIM + p) * DSTATE);
    float y = 0.0f;
    #pragma unroll 8
    for (int s = 0; s < DSTATE; ++s) {
        float v = st[s] * dA + dtx * sB[s];
        st[s] = v;
        y += v * sC[s];
    }
    y += D_param[h] * x;
    float z = zx[((long)b * L_SEQ + t) * DINPROJ + h * HEADDIM + p];
    g_buf[b * DSSM + h * HEADDIM + p] = y * siluf_(z);
}

// ---------------- per-step RMS norm -> yn (stored bf16 for next GEMMs) ------
__global__ __launch_bounds__(256) void step_norm(
    const float* __restrict__ g_buf, const float* __restrict__ norm_w,
    unsigned short* __restrict__ yn_all, int t)
{
    const int b = blockIdx.x, tid = threadIdx.x;
    __shared__ float red[256];
    const float* g = g_buf + b * DSSM;
    float s = 0.0f;
    for (int c = tid; c < DSSM; c += 256) { float v = g[c]; s += v * v; }
    red[tid] = s; __syncthreads();
    for (int off = 128; off > 0; off >>= 1) {
        if (tid < off) red[tid] += red[tid + off];
        __syncthreads();
    }
    float rs = rsqrtf(red[0] / (float)DSSM + EPSN);
    unsigned short* dst = yn_all + ((long)b * L_SEQ + t) * DSSM;
    for (int c = tid; c < DSSM; c += 256)
        dst[c] = f32_to_bf16_bits(g[c] * rs * norm_w[c]);
}

// ======================== host side ========================
extern "C" void kernel_launch(void* const* d_in, const int* in_sizes, int n_in,
                              void* d_out, int out_size, void* d_ws, size_t ws_size,
                              hipStream_t stream) {
    const float* u       = (const float*)d_in[0];
    const float* W_in    = (const float*)d_in[1];
    const float* conv_w  = (const float*)d_in[2];
    const float* conv_b  = (const float*)d_in[3];
    const float* W_rxbc  = (const float*)d_in[4];
    const float* W_rdt   = (const float*)d_in[5];
    const float* dt_bias = (const float*)d_in[6];
    const float* A_log   = (const float*)d_in[7];
    const float* D_param = (const float*)d_in[8];
    const float* norm_w  = (const float*)d_in[9];
    const float* W_out   = (const float*)d_in[10];
    float* out = (float*)d_out;

    // ---- workspace carve (256B aligned) ----
    char* ws = (char*)d_ws;
    auto carve = [&](size_t bytes) -> void* {
        void* p = ws; ws += (bytes + 255) & ~(size_t)255; return p;
    };
    const int ROWS = B_SZ * L_SEQ;                                   // 8192
    float*          zx       = (float*)         carve((size_t)ROWS * DINPROJ * 4);
    unsigned short* u16      = (unsigned short*)carve((size_t)ROWS * DMODEL * 2);
    unsigned short* win16    = (unsigned short*)carve((size_t)DINPROJ * DMODEL * 2);
    unsigned short* wr16     = (unsigned short*)carve((size_t)NREC * DSSM * 2);
    unsigned short* wout16   = (unsigned short*)carve((size_t)DMODEL * DSSM * 2);
    float*          xbc_conv = (float*)         carve((size_t)ROWS * CONVDIM * 4);
    float*          state    = (float*)         carve((size_t)B_SZ * NHEADS * HEADDIM * DSTATE * 4);
    float*          xbct     = (float*)         carve((size_t)B_SZ * NREC * 4);
    float*          g_buf    = (float*)         carve((size_t)B_SZ * DSSM * 4);
    unsigned short* yn16     = (unsigned short*)carve((size_t)ROWS * DSSM * 2);
    (void)ws_size; (void)in_sizes; (void)n_in; (void)out_size;

    // ---- 1. convert fp32 operands to bf16 ----
    auto cvt = [&](const float* s, unsigned short* d, int n) {
        cvt_f32_bf16<<<(n + 255) / 256, 256, 0, stream>>>(s, d, n);
    };
    cvt(u,      u16,                      ROWS * DMODEL);
    cvt(W_in,   win16,                    DINPROJ * DMODEL);
    cvt(W_rxbc, wr16,                     CONVDIM * DSSM);
    cvt(W_rdt,  wr16 + (size_t)CONVDIM * DSSM, NHEADS * DSSM);
    cvt(W_out,  wout16,                   DMODEL * DSSM);

    // ---- 2. input projection GEMM (8192 x 4384 = u @ W_in^T) ----
    {
        int Ntiles = DINPROJ / 16;                         // 274 (even)
        dim3 grid(ROWS / 32, (Ntiles / 2 + 3) / 4);         // (256, 35)
        wmma_gemm_bf16<<<grid, 128, 0, stream>>>(u16, DMODEL, win16, DMODEL,
                                                 zx, DINPROJ, DMODEL, Ntiles);
    }

    // ---- 3. depthwise causal conv ----
    {
        int n = ROWS * CONVDIM;
        conv1d_kernel<<<(n + 255) / 256, 256, 0, stream>>>(zx, conv_w, conv_b, xbc_conv, n);
    }

    // ---- 4. zero SSM state (deterministic each call) ----
    hipMemsetAsync(state, 0, (size_t)B_SZ * NHEADS * HEADDIM * DSTATE * 4, stream);

    // ---- 5. sequential scan over L ----
    for (int t = 0; t < L_SEQ; ++t) {
        const unsigned short* yprev = yn16 + (size_t)(t > 0 ? t - 1 : 0) * DSSM;
        step_rec_gemm<<<(NREC / 32 + 3) / 4, 128, 0, stream>>>(
            yprev, wr16, xbc_conv, zx, xbct, t, (t == 0) ? 1 : 0);
        step_state<<<dim3(NHEADS, B_SZ), 64, 0, stream>>>(
            xbct, zx, dt_bias, A_log, D_param, state, g_buf, t);
        step_norm<<<B_SZ, 256, 0, stream>>>(g_buf, norm_w, yn16, t);
    }

    // ---- 6. output projection GEMM (8192 x 1024 = yn @ W_out^T) ----
    {
        int Ntiles = DMODEL / 16;                          // 64 (even)
        dim3 grid(ROWS / 32, (Ntiles / 2 + 3) / 4);         // (256, 8)
        wmma_gemm_bf16<<<grid, 128, 0, stream>>>(yn16, DSSM, wout16, DSSM,
                                                 out, DMODEL, DSSM, Ntiles);
    }
}
